// deromconv_63617055588615
// MI455X (gfx1250) — matrix operand order repaired and checked by reading, not compiled, not run on hardware
//
#include <hip/hip_runtime.h>
#include <hip/hip_bf16.h>

// ---------------------------------------------------------------------------
// Deformable conv v2 (DCNv2) for MI455X / gfx1250, wave32 + WMMA.
//
// Implicit GEMM: M = B*H*W = 262144 pixels, K = C*9 = 576, N = 128 (main) and
// N = 27->32 (offset conv), on v_wmma_f32_16x16x32_f16 with f32 accumulation.
//
// KEY TRICK: the K dimension is traversed in a PERMUTED order kp = k2*64 + c
// (tap-major, channel-minor). Each 32-wide K-tile then touches exactly ONE
// bilinear tap (k2 = kt>>1) and a contiguous channel block (c = (kt&1)*32 +
// koff), so there is no division and no per-element tap indexing in the hot
// loop. The weight pack kernel applies the same permutation to B, keeping the
// contraction exact.
//
// WMMA fragment layouts (CDNA5 ISA 7.12.2, wave32):
//   A (16x32 f16):  lane l (0..15) -> M=l,    VGPR0..3 K=0..7,  VGPR4..7 K=16..23
//                   lane l (16..31)-> M=l-16, VGPR0..3 K=8..15, VGPR4..7 K=24..31
//   B (32x16 f16):  lane l -> N=l%16; K by lane half as in A, 2 halves/dword in VGPR0..7
//   C/D (16x16 f32):VGPR r -> M=r (lanes 0-15) / M=r+8 (lanes 16-31), N=lane%16
// ---------------------------------------------------------------------------

typedef __attribute__((ext_vector_type(16))) _Float16 v16h;
typedef __attribute__((ext_vector_type(8)))  float    v8f;

constexpr int Hc   = 128;
constexpr int Wc   = 128;
constexpr int Cc   = 64;
constexpr int Oc   = 128;
constexpr int HW   = Hc * Wc;      // 16384
constexpr int KTOT = Cc * 9;       // 576
constexpr int NKT  = KTOT / 32;    // 18 K-tiles

// A-layout K offset for element e (0..15), low half-wave (hi adds 8).
__device__ __forceinline__ constexpr int a_koff_lo(int e) {
    return e + ((e & 8) ? 8 : 0);
}

// Load one pre-packed B fragment: lane reads its 8 contiguous dwords (2x b128).
__device__ __forceinline__ v16h load_bfrag(const int4* __restrict__ packed,
                                           int frag, int lane) {
    union { int4 q[2]; v16h h; } u;
    const int4* p = packed + ((size_t)frag * 32 + lane) * 2;
    u.q[0] = p[0];
    u.q[1] = p[1];
    return u.h;
}

// ---------------------------------------------------------------------------
// Weight packing with the PERMUTED K order: kp = kt*32 + pair*16 + (lane>>4)*8 + j,
// original k = (kp & 63) * 9 + (kp >> 6).  src is [N][576] row-major f32.
// ---------------------------------------------------------------------------
__global__ void pack_weights(const float* __restrict__ src, int Nvalid, int NT,
                             int4* __restrict__ dst) {
    int idx   = blockIdx.x * 256 + threadIdx.x;
    int total = NKT * NT * 32 * 2;
    if (idx >= total) return;
    int pair = idx & 1;          // which int4 of the lane's pair
    int lane = (idx >> 1) & 31;  // B column lane&15, K half lane>>4
    int frag = idx >> 6;         // kt*NT + nt
    int kt   = frag / NT;
    int nt   = frag - kt * NT;
    int kpb  = kt * 32 + pair * 16 + (lane >> 4) * 8;  // permuted K base
    int n    = nt * 16 + (lane & 15);
    union { int4 v; _Float16 h[8]; } u;
#pragma unroll
    for (int j = 0; j < 8; ++j) {
        int kp = kpb + j;
        int k  = (kp & 63) * 9 + (kp >> 6);  // original K index c*9 + k2
        float val = (n < Nvalid) ? src[(size_t)n * KTOT + k] : 0.f;
        u.h[j] = (_Float16)val;
    }
    dst[(size_t)frag * 64 + lane * 2 + pair] = u.v;
}

// ---------------------------------------------------------------------------
// Fused kernel: 4 waves/block, each wave owns one 16-pixel M-tile.
// ---------------------------------------------------------------------------
__global__ __launch_bounds__(128)
void deform_fused(const float* __restrict__ x,
                  const float* __restrict__ b_off,
                  const float* __restrict__ b_def,
                  const int4*  __restrict__ pOff,
                  const int4*  __restrict__ pDef,
                  float* __restrict__ out) {
    __shared__ float offStage[4][16][32];  // per-wave 16 pixels x 32(27) chans
    __shared__ int4  taps[4][16][9];       // y0|x0 i16, (w00,w01) f16, (w10,w11) f16

    const int tid     = threadIdx.x;
    const int wave    = tid >> 5;
    const int lane    = tid & 31;
    const int mrow    = lane & 15;
    const int hi      = lane >> 4;         // half-wave index
    const int mtile   = blockIdx.x * 4 + wave;
    const int pixbase = mtile * 16;

    // This lane's A-row pixel
    const int p  = pixbase + mrow;
    const int bb = p >> 14;                // / HW
    const int hh = (p >> 7) & 127;
    const int ww = p & 127;
    // base with the half-wave channel offset (hi lanes start at c = +8)
    const float* xbh = x + (size_t)bb * (Cc * HW) + ((size_t)hi << 17);

    v8f zero = {};

    // ---------------- Phase 1: offset conv (N = 27 padded to 32) -----------
    v8f acc0 = zero, acc1 = zero;
    {
        int k2 = 0;
#pragma unroll 1
        for (int ky = 0; ky < 3; ++ky) {
#pragma unroll 1
            for (int kx = 0; kx < 3; ++kx, ++k2) {
                int y  = hh + ky - 1;
                int xw = ww + kx - 1;
                bool valid = ((unsigned)y < (unsigned)Hc) &&
                             ((unsigned)xw < (unsigned)Wc);
                int yc = min(max(y, 0), Hc - 1);
                int xc = min(max(xw, 0), Wc - 1);
                float pm = valid ? 1.f : 0.f;
                const float* base = xbh + ((yc << 7) + xc);
#pragma unroll
                for (int half = 0; half < 2; ++half) {
                    const float* bh = base + ((size_t)half << 19); // c += 32
                    v16h a;
#pragma unroll
                    for (int e = 0; e < 16; ++e) {
                        float v = bh[a_koff_lo(e) << 14] * pm;
                        a[e] = (_Float16)v;
                    }
                    int kt = k2 * 2 + half;
                    v16h bf0 = load_bfrag(pOff, kt * 2 + 0, lane);
                    acc0 = __builtin_amdgcn_wmma_f32_16x16x32_f16(
                        false, a, false, bf0, (short)0, acc0, false, false);
                    v16h bf1 = load_bfrag(pOff, kt * 2 + 1, lane);
                    acc1 = __builtin_amdgcn_wmma_f32_16x16x32_f16(
                        false, a, false, bf1, (short)0, acc1, false, false);
                }
            }
        }
    }
    {
        int n0 = mrow;
        int n1 = 16 + mrow;
        float bo0 = b_off[n0];
        float bo1 = (n1 < 27) ? b_off[n1] : 0.f;
#pragma unroll
        for (int r = 0; r < 8; ++r) {
            offStage[wave][r + 8 * hi][n0] = acc0[r] + bo0;
            offStage[wave][r + 8 * hi][n1] = acc1[r] + bo1;
        }
    }
    __syncthreads();

    // ---------------- Phase 1.5: bilinear tap metadata ----------------------
    // offset channels: 2*k2 = dy, 2*k2+1 = dx, 18+k2 = mask logit.
    for (int t = lane; t < 16 * 9; t += 32) {
        int m  = t / 9;
        int k2 = t - 9 * m;
        float dy   = offStage[wave][m][2 * k2];
        float dx   = offStage[wave][m][2 * k2 + 1];
        float mlog = offStage[wave][m][18 + k2];
        float msk  = 1.f / (1.f + expf(-mlog));
        int pp = pixbase + m;
        int ph = (pp >> 7) & 127;
        int pw = pp & 127;
        int ky = k2 / 3;
        int kx = k2 - 3 * ky;
        float py = dy + (float)(ph + ky - 1);
        float px = dx + (float)(pw + kx - 1);
        float fy = floorf(py), fx = floorf(px);
        int   y0 = (int)fy,    x0 = (int)fx;
        float wy = py - fy,    wx = px - fx;
        bool vy0 = (y0 >= 0)  && (y0 < Hc);
        bool vy1 = (y0 >= -1) && (y0 < Hc - 1);
        bool vx0 = (x0 >= 0)  && (x0 < Wc);
        bool vx1 = (x0 >= -1) && (x0 < Wc - 1);
        float w00 = (1.f - wy) * (1.f - wx) * msk; if (!(vy0 && vx0)) w00 = 0.f;
        float w01 = (1.f - wy) * wx         * msk; if (!(vy0 && vx1)) w01 = 0.f;
        float w10 = wy         * (1.f - wx) * msk; if (!(vy1 && vx0)) w10 = 0.f;
        float w11 = wy         * wx         * msk; if (!(vy1 && vx1)) w11 = 0.f;
        y0 = min(max(y0, -2), Hc + 1);  // weights already zeroed if OOB
        x0 = min(max(x0, -2), Wc + 1);
        int4 tp;
        tp.x = (y0 & 0xffff) | (x0 << 16);
        union { int i; _Float16 h[2]; } u;
        u.h[0] = (_Float16)w00; u.h[1] = (_Float16)w01; tp.y = u.i;
        u.h[0] = (_Float16)w10; u.h[1] = (_Float16)w11; tp.z = u.i;
        tp.w = 0;
        taps[wave][m][k2] = tp;
    }
    __syncthreads();

    // ---------------- Phase 2: main deformable GEMM (N = 128) --------------
    v8f acc[8];
#pragma unroll
    for (int i = 0; i < 8; ++i) acc[i] = zero;

#pragma unroll 1
    for (int k2 = 0; k2 < 9; ++k2) {
        // One tap per K-tile pair: load from LDS and decode once.
        int4 tp = taps[wave][mrow][k2];
        int y0 = (int)(short)(tp.x & 0xffff);
        int x0 = tp.x >> 16;  // arithmetic shift: sign-extended
        int y0c = min(max(y0,     0), Hc - 1);
        int y1c = min(max(y0 + 1, 0), Hc - 1);
        int x0c = min(max(x0,     0), Wc - 1);
        int x1c = min(max(x0 + 1, 0), Wc - 1);
        union { int i; _Float16 h[2]; } uA, uB;
        uA.i = tp.y; uB.i = tp.z;
        float w00 = (float)uA.h[0], w01 = (float)uA.h[1];
        float w10 = (float)uB.h[0], w11 = (float)uB.h[1];
        const float* b00 = xbh + ((y0c << 7) + x0c);
        const float* b01 = xbh + ((y0c << 7) + x1c);
        const float* b10 = xbh + ((y1c << 7) + x0c);
        const float* b11 = xbh + ((y1c << 7) + x1c);
#pragma unroll
        for (int half = 0; half < 2; ++half) {
            const size_t hoff = (size_t)half << 19;  // c += 32
            v16h a;
#pragma unroll
            for (int e = 0; e < 16; ++e) {
                const int co = a_koff_lo(e) << 14;   // compile-time immediate
                float g00 = b00[hoff + co];
                float g01 = b01[hoff + co];
                float g10 = b10[hoff + co];
                float g11 = b11[hoff + co];
                float v = w00 * g00 + w01 * g01 + w10 * g10 + w11 * g11;
                a[e] = (_Float16)v;
            }
            int kt = k2 * 2 + half;
#pragma unroll
            for (int nt = 0; nt < 8; ++nt) {
                v16h bf = load_bfrag(pDef, kt * 8 + nt, lane);
                acc[nt] = __builtin_amdgcn_wmma_f32_16x16x32_f16(
                    false, a, false, bf, (short)0, acc[nt], false, false);
            }
        }
    }

    // ---------------- Epilogue: bias + scatter store (NCHW) -----------------
#pragma unroll
    for (int nt = 0; nt < 8; ++nt) {
        int n = nt * 16 + mrow;
        float bd = b_def[n];
#pragma unroll
        for (int r = 0; r < 8; ++r) {
            int m   = r + 8 * hi;
            int pp  = pixbase + m;
            int ob  = pp >> 14;
            int ohw = pp & 16383;
            out[((size_t)(ob * Oc + n) << 14) + ohw] = acc[nt][r] + bd;
        }
    }
}

// ---------------------------------------------------------------------------
extern "C" void kernel_launch(void* const* d_in, const int* in_sizes, int n_in,
                              void* d_out, int out_size, void* d_ws, size_t ws_size,
                              hipStream_t stream) {
    const float* x     = (const float*)d_in[0];
    const float* w_off = (const float*)d_in[1];
    const float* b_off = (const float*)d_in[2];
    const float* w_def = (const float*)d_in[3];
    const float* b_def = (const float*)d_in[4];
    float* out = (float*)d_out;

    // Workspace: packed w_off fragments (36 KB), then w_def fragments (144 KB).
    int4* pOff = (int4*)d_ws;
    int4* pDef = pOff + (size_t)NKT * 2 * 64;

    {
        int total = NKT * 2 * 32 * 2;  // 2304 threads
        pack_weights<<<(total + 255) / 256, 256, 0, stream>>>(w_off, 27, 2, pOff);
    }
    {
        int total = NKT * 8 * 32 * 2;  // 9216 threads
        pack_weights<<<(total + 255) / 256, 256, 0, stream>>>(w_def, 128, 8, pDef);
    }

    // 262144 pixels / (16 per M-tile * 4 waves per block) = 4096 blocks.
    deform_fused<<<4096, 128, 0, stream>>>(x, b_off, b_def, pOff, pDef, out);
}
